// AutoCorrelation_19224273617548
// MI455X (gfx1250) — matrix operand order repaired and checked by reading, compiled, not run on hardware
//
#include <hip/hip_runtime.h>
#include <hip/hip_bf16.h>

typedef __attribute__((ext_vector_type(16))) __bf16 v16bf;
typedef __attribute__((ext_vector_type(8)))  __bf16 v8bf;
typedef __attribute__((ext_vector_type(8)))  float  v8f;

#define BD   16
#define LD   2048
#define HD   16
#define ED   64
#define TOPK 22          // int(3 * ln(2048)) = 22
#define NBH  (BD * HD)

// ---------------------------------------------------------------- zero ws ---
__global__ void k_zero(float* __restrict__ p, int n) {
    int i = blockIdx.x * blockDim.x + threadIdx.x;
    if (i < n) p[i] = 0.0f;
}

// ---------------------------------------------- mean autocorrelation (WMMA) -
// Grid (NBH, 2): each workgroup owns one (b,h) and 64 of the 128 block-shifts.
// Q[2048][64] staged as bf16 in LDS (256 KB of the 320 KB WGP LDS).
// Per wave: 8 live accumulators T_Delta; A_i fragment hoisted and reused by
// all 8 shifts -> 36 ds_load_b128 per 16 WMMAs (vs 64 without reuse).
__global__ void __launch_bounds__(256)
k_corr(const float* __restrict__ qk, float* __restrict__ corr) {
    __shared__ __bf16 sQ[LD * ED];           // 262144 B

    const int bh = blockIdx.x;
    const int b  = bh / HD, h = bh % HD;
    const float* __restrict__ qbase = qk + (size_t)b * LD * HD * ED + (size_t)h * ED;

    // stage + f32 -> bf16 convert (coalesced: 64 contiguous floats per row)
    for (int idx = threadIdx.x; idx < LD * ED; idx += blockDim.x) {
        int l = idx >> 6, e = idx & 63;
        sQ[idx] = (__bf16)qbase[(size_t)l * HD * ED + e];
    }
    __syncthreads();

    const int lane = threadIdx.x & 31;
    const int half = lane >> 4;              // wave32 lane half
    const int mn   = lane & 15;              // M for A-frag, N for B-frag
    const int wave = threadIdx.x >> 5;       // 8 waves per WG
    const int dz   = blockIdx.y * 64;        // this WG's shift range base

    // A-matrix 16x32 bf16 fragment: lane m = lane%16,
    // VGPR pairs cover K = {0..7, 16..23} (half 0) / {8..15, 24..31} (half 1)
    auto load_fragA = [&](int row, int kbase) -> v16bf {
        v8bf lo = *(const v8bf*)&sQ[row * ED + kbase + 8 * half];        // K +0..7
        v8bf hi = *(const v8bf*)&sQ[row * ED + kbase + 16 + 8 * half];   // K +16..23
        v16bf f;
        #pragma unroll
        for (int j = 0; j < 8; ++j) { f[j] = lo[j]; f[8 + j] = hi[j]; }
        return f;
    };
    // B-matrix 32x16 bf16 fragment: lane n = lane%16, rows K = {0..7, 16..23} (half 0) / {8..15, 24..31} (half 1).
    // B = A_j^T, so B[k][n] = Q[rowJ + n][k] -> two contiguous 8-bf16 runs along k.
    auto load_fragB = [&](int row, int kbase) -> v16bf {
        v8bf lo = *(const v8bf*)&sQ[row * ED + kbase + 8 * half];
        v8bf hi = *(const v8bf*)&sQ[row * ED + kbase + 16 + 8 * half];
        v16bf f;
        #pragma unroll
        for (int j = 0; j < 8; ++j) { f[j] = lo[j]; f[8 + j] = hi[j]; }
        return f;
    };

    // wave owns Delta = dz + wave + 8*d, d = 0..7  (8 independent chains)
    v8f acc[8];
    #pragma unroll
    for (int d = 0; d < 8; ++d) acc[d] = (v8f){};

    for (int i = 0; i < 128; ++i) {
        const int rowI = i * 16 + mn;                       // A: M = mn
        const v16bf a0 = load_fragA(rowI, 0);               // reused by 8 shifts
        const v16bf a1 = load_fragA(rowI, 32);
        #pragma unroll
        for (int d = 0; d < 8; ++d) {
            const int delta = dz + wave + 8 * d;
            const int rowJ  = ((i + delta) & 127) * 16 + mn;  // B: N = mn
            v16bf b0 = load_fragB(rowJ, 0);
            acc[d] = __builtin_amdgcn_wmma_f32_16x16x32_bf16(
                         false, a0, false, b0, (short)0, acc[d], false, false);
            v16bf b1 = load_fragB(rowJ, 32);
            acc[d] = __builtin_amdgcn_wmma_f32_16x16x32_bf16(
                         false, a1, false, b1, (short)0, acc[d], false, false);
        }
    }

    // C/D layout: element (M = p + 8*half, N = mn) in acc[d][p]
    // T_Delta[m][n] contributes to tau = (16*Delta + n - m) mod L
    #pragma unroll
    for (int d = 0; d < 8; ++d) {
        const int delta = dz + wave + 8 * d;
        #pragma unroll
        for (int p = 0; p < 8; ++p) {
            int m   = p + 8 * half;
            int tau = (16 * delta + mn - m) & (LD - 1);
            atomicAdd(&corr[(size_t)bh * LD + tau], acc[d][p]);
        }
    }
}

// --------------------------------------------------- top-22 + softmax -------
__global__ void __launch_bounds__(256)
k_topk(const float* __restrict__ corr, float* __restrict__ w, int* __restrict__ d) {
    __shared__ float sv[LD];
    __shared__ float rv[256];
    __shared__ int   ri[256];
    __shared__ float tw[TOPK];
    __shared__ int   td[TOPK];

    const int bh = blockIdx.x, t = threadIdx.x;
    for (int i = t; i < LD; i += 256)
        sv[i] = corr[(size_t)bh * LD + i] * (1.0f / ED);   // mean over E
    __syncthreads();

    for (int k = 0; k < TOPK; ++k) {
        float best = -3.4e38f; int bi = 0;
        for (int i = t; i < LD; i += 256) {
            float v = sv[i];
            if (v > best) { best = v; bi = i; }
        }
        rv[t] = best; ri[t] = bi;
        __syncthreads();
        for (int s = 128; s > 0; s >>= 1) {
            if (t < s && rv[t + s] > rv[t]) { rv[t] = rv[t + s]; ri[t] = ri[t + s]; }
            __syncthreads();
        }
        if (t == 0) { tw[k] = rv[0]; td[k] = ri[0]; sv[ri[0]] = -3.4e38f; }
        __syncthreads();
    }

    if (t == 0) {
        float mx = tw[0], s = 0.0f;     // tw[0] is the global max
        float ex[TOPK];
        for (int k = 0; k < TOPK; ++k) { ex[k] = __expf(tw[k] - mx); s += ex[k]; }
        float inv = 1.0f / s;
        for (int k = 0; k < TOPK; ++k) {
            w[bh * TOPK + k] = ex[k] * inv;
            d[bh * TOPK + k] = td[k];
        }
    }
}

// ------------------------------------- weighted circular gather (L2-resident)
__global__ void __launch_bounds__(512)
k_agg(const float* __restrict__ vals, const float* __restrict__ w,
      const int* __restrict__ d, float* __restrict__ out) {
    __shared__ float sw[TOPK];
    __shared__ int   sd[TOPK];

    const int gid = blockIdx.x;
    const int bh  = gid >> 8;            // 256 chunks of 8 rows per (b,h)
    const int l0  = (gid & 255) * 8;
    const int b   = bh / HD, h = bh % HD;

    if (threadIdx.x < TOPK) {
        sw[threadIdx.x] = w[bh * TOPK + threadIdx.x];
        sd[threadIdx.x] = d[bh * TOPK + threadIdx.x];
    }
    __syncthreads();

    const int e = threadIdx.x & 63;
    const int l = l0 + (threadIdx.x >> 6);
    const float* __restrict__ vb = vals + (size_t)b * LD * HD * ED + (size_t)h * ED;

    float acc = 0.0f;
    #pragma unroll
    for (int k = 0; k < TOPK; ++k) {
        int lp = (l + sd[k]) & (LD - 1);
        acc += sw[k] * vb[(size_t)lp * HD * ED + e];
    }
    out[(size_t)b * LD * HD * ED + (size_t)l * HD * ED + (size_t)h * ED + e] = acc;
}

// ---------------------------------------------------------------------------
extern "C" void kernel_launch(void* const* d_in, const int* in_sizes, int n_in,
                              void* d_out, int out_size, void* d_ws, size_t ws_size,
                              hipStream_t stream) {
    const float* qk   = (const float*)d_in[0];
    const float* vals = (const float*)d_in[1];
    float* out  = (float*)d_out;

    float* corr = (float*)d_ws;                    // [NBH * LD] f32   (2 MB)
    float* w    = corr + (size_t)NBH * LD;         // [NBH * TOPK] f32
    int*   dl   = (int*)(w + (size_t)NBH * TOPK);  // [NBH * TOPK] i32

    const int n = NBH * LD;
    k_zero<<<(n + 511) / 512, 512, 0, stream>>>(corr, n);
    k_corr<<<dim3(NBH, 2), 256, 0, stream>>>(qk, corr);
    k_topk<<<NBH, 256, 0, stream>>>(corr, w, dl);
    k_agg <<<NBH * 256, 512, 0, stream>>>(vals, w, dl, out);
}
